// SelfAttnE2V_43336220017261
// MI455X (gfx1250) — compile-verified, matrix operand order checked by reading, not run on hardware
//
#include <hip/hip_runtime.h>
#include <hip/hip_bf16.h>
#include <cstddef>

// ---------------------------------------------------------------------------
// Types for WMMA fragments (wave32, v_wmma_f32_16x16x32_f16)
// ---------------------------------------------------------------------------
typedef __attribute__((ext_vector_type(16))) _Float16 v16h;
typedef __attribute__((ext_vector_type(8)))  _Float16 v8h;
typedef __attribute__((ext_vector_type(8)))  float    v8f;

union Frag { v16h f; v8h h[2]; };

#define DIMD 256
#define HIDW 512

// ---------------------------------------------------------------------------
// WMMA GEMM:  C[M,Ncol] = A[M,K](f16,row) @ Bt[Ncol,K](f16,row)^T
// Wave tile = (MT*16) x (NT*16); each B fragment is reused across MT M-tiles
// (MT=2,NT=4: 12 b128 loads per 8 WMMAs vs 10 per 4 at MT=1).
// The k-loop is branch-free: A row indices are clamped (not branched) for the
// M-tail, and only epilogue stores are guarded, so accumulators stay pinned.
// A-fragment (16-bit A 16x32): lane L: g=L/16, m=L%16;
//   halves 0..7  = A[m][k0 + g*8 .. +7]      (contiguous 16B)
//   halves 8..15 = A[m][k0 + 16 + g*8 .. +7] (contiguous 16B)
// Bt rows use the symmetric layout -> same two contiguous 16B loads per lane.
// ---------------------------------------------------------------------------
template<int MT, int NT, bool RELU>
__global__ void __launch_bounds__(256)
gemm_wmma_kernel(const _Float16* __restrict__ A,
                 const _Float16* __restrict__ Bt,
                 int M, int K, int Ncol,
                 const float* __restrict__ bias1,
                 const float* __restrict__ bias2,
                 const float* __restrict__ residA,
                 const float* __restrict__ residB,
                 int split,
                 _Float16* __restrict__ outH,
                 float* __restrict__ outF)
{
    const int wave = threadIdx.x >> 5;
    const int lane = threadIdx.x & 31;
    const int g    = lane >> 4;
    const int lr   = lane & 15;
    const int m0   = (blockIdx.x * 8 + wave) * (MT * 16);
    const int n0   = blockIdx.y * (NT * 16);
    if (m0 >= M) return;

    v8f acc[MT][NT] = {};

    // raw pointers stepped by 32 halves per iteration; A rows clamped for tail
    const _Float16* ap0[MT];
    const _Float16* ap1[MT];
#pragma unroll
    for (int mt = 0; mt < MT; ++mt) {
        int rr = m0 + mt * 16 + lr;
        rr = rr < M ? rr : (M - 1);          // branchless tail clamp (loads only)
        ap0[mt] = A + (size_t)rr * K + g * 8;
        ap1[mt] = ap0[mt] + 16;
    }
    const _Float16* bp0[NT];
    const _Float16* bp1[NT];
#pragma unroll
    for (int t = 0; t < NT; ++t) {
        bp0[t] = Bt + (size_t)(n0 + t * 16 + lr) * K + g * 8;
        bp1[t] = bp0[t] + 16;
    }

    for (int k0 = 0; k0 < K; k0 += 32) {
        Frag a[MT];
#pragma unroll
        for (int mt = 0; mt < MT; ++mt) {
            a[mt].h[0] = *(const v8h*)ap0[mt];
            a[mt].h[1] = *(const v8h*)ap1[mt];
            ap0[mt] += 32; ap1[mt] += 32;
        }
#pragma unroll
        for (int t = 0; t < NT; ++t) {
            Frag b;
            b.h[0] = *(const v8h*)bp0[t];
            b.h[1] = *(const v8h*)bp1[t];
            bp0[t] += 32; bp1[t] += 32;
#pragma unroll
            for (int mt = 0; mt < MT; ++mt)
                acc[mt][t] = __builtin_amdgcn_wmma_f32_16x16x32_f16(
                    false, a[mt].f, false, b.f, (short)0, acc[mt][t], false, false);
        }
    }

    // C layout: lane L: n = L%16 (=lr), VGPR j -> row base + (L/16)*8 + j
#pragma unroll
    for (int mt = 0; mt < MT; ++mt) {
#pragma unroll
        for (int t = 0; t < NT; ++t) {
            const int col = n0 + t * 16 + lr;
#pragma unroll
            for (int j = 0; j < 8; ++j) {
                const int row = m0 + mt * 16 + g * 8 + j;
                if (row >= M) continue;      // tail guard (stores only)
                float s = acc[mt][t][j];
                if (bias1) s += bias1[col];
                if (bias2) s += bias2[col];
                if (residA) {
                    if (row < split) s += residA[(size_t)row * Ncol + col];
                    else             s += residB[(size_t)(row - split) * Ncol + col];
                }
                if (RELU) s = s > 0.f ? s : 0.f;
                const size_t o = (size_t)row * Ncol + col;
                if (outH) outH[o] = (_Float16)s;
                if (outF) outF[o] = s;
            }
        }
    }
}

// ---------------------------------------------------------------------------
// Weight prep: float W[K,Ncol] -> half Wt[Ncol,K]
// ---------------------------------------------------------------------------
__global__ void __launch_bounds__(256)
w2h_t_kernel(const float* __restrict__ W, _Float16* __restrict__ Wt, int K, int Ncol)
{
    int t = blockIdx.x * 256 + threadIdx.x;
    if (t >= K * Ncol) return;
    int n = t / K, k = t % K;
    Wt[t] = (_Float16)W[(size_t)k * Ncol + n];
}

// ---------------------------------------------------------------------------
// Tiny q MLP: q[2,256] = relu(peq[2,128]@w1 + b1)@w2 + b2
// ---------------------------------------------------------------------------
__global__ void __launch_bounds__(256)
qmlp_kernel(const float* __restrict__ peq,
            const float* __restrict__ w1, const float* __restrict__ b1,
            const float* __restrict__ w2, const float* __restrict__ b2,
            float* __restrict__ qout)
{
    const int r = blockIdx.x;   // 0..1
    const int t = threadIdx.x;  // 0..255
    __shared__ float inp[128];
    __shared__ float hid[HIDW];
    if (t < 128) inp[t] = peq[r * 128 + t];
    __syncthreads();
    for (int j = t; j < HIDW; j += 256) {
        float s = b1[j];
        for (int i = 0; i < 128; ++i) s += inp[i] * w1[i * HIDW + j];
        hid[j] = s > 0.f ? s : 0.f;
    }
    __syncthreads();
    float s = b2[t];
    for (int j = 0; j < HIDW; ++j) s += hid[j] * w2[j * DIMD + t];
    qout[r * DIMD + t] = s;
}

// ---------------------------------------------------------------------------
// Fold q into k_w: Wlogt[16][256] (half), blog[16].
// logit[e,h] = xs[e] . Wlog[:,h] + blog[h], scale 1/sqrt(32) folded in.
// h<8 : q row 0, k_w cols [h*32, h*32+32)      (logit0)
// h>=8: q row 1, k_w cols [256+(h-8)*32, ...)  (logit1)
// ---------------------------------------------------------------------------
__global__ void __launch_bounds__(256)
wlogit_kernel(const float* __restrict__ k_w, const float* __restrict__ k_b,
              const float* __restrict__ q,
              _Float16* __restrict__ Wlogt, float* __restrict__ blog)
{
    int t = blockIdx.x * 256 + threadIdx.x;
    if (t >= 16 * DIMD) return;
    const int h = t / DIMD, d = t % DIMD;
    const float scale = 0.17677669529663687f; // 1/sqrt(32)
    const int qrow = (h < 8) ? 0 : 1;
    const int off  = (h < 8) ? h * 32 : DIMD + (h - 8) * 32;
    const int qoff = (h < 8) ? h * 32 : (h - 8) * 32;
    float s = 0.f;
    for (int j = 0; j < 32; ++j)
        s += k_w[(size_t)d * (2 * DIMD) + off + j] * q[qrow * DIMD + qoff + j];
    Wlogt[h * DIMD + d] = (_Float16)(s * scale);
    if (d == 0) {
        float bb = 0.f;
        for (int j = 0; j < 32; ++j) bb += k_b[off + j] * q[qrow * DIMD + qoff + j];
        blog[h] = bb * scale;
    }
}

// ---------------------------------------------------------------------------
// LayerNorm + concat(PE) -> half [rows, 512]. One block per row.
// ---------------------------------------------------------------------------
__global__ void __launch_bounds__(256)
build_cat_kernel(const float* __restrict__ xA, const float* __restrict__ xB, int split,
                 const int* __restrict__ orders, int fixed_ord,
                 const float* __restrict__ pe,
                 const float* __restrict__ g, const float* __restrict__ b,
                 _Float16* __restrict__ cat)
{
    const int row = blockIdx.x;
    const int d   = threadIdx.x;
    const float* x = (row < split) ? (xA + (size_t)row * DIMD)
                                   : (xB + (size_t)(row - split) * DIMD);
    float v = x[d];
    __shared__ float red[256];
    red[d] = v; __syncthreads();
    for (int s = 128; s > 0; s >>= 1) { if (d < s) red[d] += red[d + s]; __syncthreads(); }
    float mean = red[0] * (1.f / 256.f);
    __syncthreads();
    float c = v - mean;
    red[d] = c * c; __syncthreads();
    for (int s = 128; s > 0; s >>= 1) { if (d < s) red[d] += red[d + s]; __syncthreads(); }
    float var = red[0] * (1.f / 256.f);
    float y = c * rsqrtf(var + 1e-5f) * g[d] + b[d];
    int ord = (orders && row < split) ? orders[row] : fixed_ord;
    cat[(size_t)row * (2 * DIMD) + d]        = (_Float16)y;
    cat[(size_t)row * (2 * DIMD) + DIMD + d] = (_Float16)pe[(size_t)ord * DIMD + d];
}

// LayerNorm only -> half [rows, 256]
__global__ void __launch_bounds__(256)
build_ln_kernel(const float* __restrict__ x,
                const float* __restrict__ g, const float* __restrict__ b,
                _Float16* __restrict__ out)
{
    const int row = blockIdx.x;
    const int d   = threadIdx.x;
    float v = x[(size_t)row * DIMD + d];
    __shared__ float red[256];
    red[d] = v; __syncthreads();
    for (int s = 128; s > 0; s >>= 1) { if (d < s) red[d] += red[d + s]; __syncthreads(); }
    float mean = red[0] * (1.f / 256.f);
    __syncthreads();
    float c = v - mean;
    red[d] = c * c; __syncthreads();
    for (int s = 128; s > 0; s >>= 1) { if (d < s) red[d] += red[d + s]; __syncthreads(); }
    float var = red[0] * (1.f / 256.f);
    out[(size_t)row * DIMD + d] = (_Float16)(c * rsqrtf(var + 1e-5f) * g[d] + b[d]);
}

// ---------------------------------------------------------------------------
// att0 (global softmax)
// ---------------------------------------------------------------------------
__global__ void __launch_bounds__(256)
att0_max_kernel(const float* __restrict__ logits, int R, float* __restrict__ pmax)
{
    const int t = threadIdx.x, h = t & 7, rl = t >> 3;
    float m = -3.4e38f;
    for (int r = blockIdx.x * 32 + rl; r < R; r += gridDim.x * 32)
        m = fmaxf(m, logits[(size_t)r * 16 + h]);
    __shared__ float sm[256];
    sm[t] = m; __syncthreads();
    for (int s = 128; s >= 8; s >>= 1) { if (t < s) sm[t] = fmaxf(sm[t], sm[t + s]); __syncthreads(); }
    if (t < 8) pmax[blockIdx.x * 8 + t] = sm[t];
}

__global__ void __launch_bounds__(256)
att0_maxfin_kernel(const float* __restrict__ pmax, int nb, float* __restrict__ gmax)
{
    const int t = threadIdx.x, h = t & 7, rl = t >> 3;
    float m = -3.4e38f;
    for (int i = rl; i < nb; i += 32) m = fmaxf(m, pmax[i * 8 + h]);
    __shared__ float sm[256];
    sm[t] = m; __syncthreads();
    for (int s = 128; s >= 8; s >>= 1) { if (t < s) sm[t] = fmaxf(sm[t], sm[t + s]); __syncthreads(); }
    if (t < 8) gmax[t] = sm[t];
}

__global__ void __launch_bounds__(256)
att0_acc_kernel(const float* __restrict__ logits, const float* __restrict__ v,
                const float* __restrict__ gmax, int R,
                float* __restrict__ pacc, float* __restrict__ psum)
{
    const int d = threadIdx.x, h = d >> 5;
    const float mh = gmax[h];
    float acc = 0.f, se = 0.f;
    for (int r = blockIdx.x; r < R; r += gridDim.x) {
        float w = __expf(logits[(size_t)r * 16 + h] - mh);
        acc += w * v[(size_t)r * DIMD + d];
        if ((d & 31) == 0) se += w;
    }
    pacc[(size_t)blockIdx.x * DIMD + d] = acc;
    if ((d & 31) == 0) psum[blockIdx.x * 8 + h] = se;
}

__global__ void __launch_bounds__(256)
att0_fin_kernel(const float* __restrict__ pacc, const float* __restrict__ psum,
                int nb, float* __restrict__ att0raw)
{
    __shared__ float S[8];
    const int d = threadIdx.x;
    if (d < 8) {
        float s = 0.f;
        for (int i = 0; i < nb; ++i) s += psum[i * 8 + d];
        S[d] = s;
    }
    __syncthreads();
    float a = 0.f;
    for (int i = 0; i < nb; ++i) a += pacc[(size_t)i * DIMD + d];
    att0raw[d] = a / S[d >> 5];
}

// single-row m2 MLP: att0fin = att0raw + mlp(concat(ln2(att0raw), pe2[0]))
__global__ void __launch_bounds__(256)
att0_mlp_kernel(const float* __restrict__ a0, const float* __restrict__ pe2r0,
                const float* __restrict__ g, const float* __restrict__ b,
                const float* __restrict__ w1, const float* __restrict__ b1,
                const float* __restrict__ w2, const float* __restrict__ b2,
                float* __restrict__ out)
{
    __shared__ float cat[2 * DIMD];
    __shared__ float hid[HIDW];
    __shared__ float red[256];
    const int t = threadIdx.x;
    float x = a0[t];
    red[t] = x; __syncthreads();
    for (int s = 128; s > 0; s >>= 1) { if (t < s) red[t] += red[t + s]; __syncthreads(); }
    float mean = red[0] * (1.f / 256.f);
    __syncthreads();
    float c = x - mean;
    red[t] = c * c; __syncthreads();
    for (int s = 128; s > 0; s >>= 1) { if (t < s) red[t] += red[t + s]; __syncthreads(); }
    float var = red[0] * (1.f / 256.f);
    cat[t]        = c * rsqrtf(var + 1e-5f) * g[t] + b[t];
    cat[DIMD + t] = pe2r0[t];
    __syncthreads();
    for (int j = t; j < HIDW; j += 256) {
        float s = b1[j];
        for (int i = 0; i < 2 * DIMD; ++i) s += cat[i] * w1[i * HIDW + j];
        hid[j] = s > 0.f ? s : 0.f;
    }
    __syncthreads();
    float s = b2[t];
    for (int j = 0; j < HIDW; ++j) s += hid[j] * w2[j * DIMD + t];
    out[t] = a0[t] + s;
}

// ---------------------------------------------------------------------------
// Sparse segment softmax (order-1 attention)
// ---------------------------------------------------------------------------
__device__ __forceinline__ unsigned fkey(float f) {
    unsigned u = __float_as_uint(f);
    return (u & 0x80000000u) ? ~u : (u | 0x80000000u);
}
__device__ __forceinline__ float funkey(unsigned k) {
    unsigned u = (k & 0x80000000u) ? (k & 0x7fffffffu) : ~k;
    return __uint_as_float(u);
}

__global__ void __launch_bounds__(256)
seg_max_kernel(const float* __restrict__ logits, const int* __restrict__ node_idx,
               const int* __restrict__ src_idx, int nnz, unsigned* __restrict__ segkey)
{
    int t = blockIdx.x * 256 + threadIdx.x;
    if (t >= nnz * 8) return;
    int nz = t >> 3, h = t & 7;
    float val = logits[(size_t)src_idx[nz] * 16 + 8 + h];
    atomicMax(&segkey[(size_t)node_idx[nz] * 8 + h], fkey(val));
}

__global__ void __launch_bounds__(256)
seg_exp_kernel(const float* __restrict__ logits, const int* __restrict__ node_idx,
               const int* __restrict__ src_idx, int nnz,
               const unsigned* __restrict__ segkey,
               float* __restrict__ ex, float* __restrict__ segsum)
{
    int t = blockIdx.x * 256 + threadIdx.x;
    if (t >= nnz * 8) return;
    int nz = t >> 3, h = t & 7;
    int node = node_idx[nz];
    float mx = funkey(segkey[(size_t)node * 8 + h]);
    float e = __expf(logits[(size_t)src_idx[nz] * 16 + 8 + h] - mx);
    ex[t] = e;
    atomicAdd(&segsum[(size_t)node * 8 + h], e);
}

__global__ void __launch_bounds__(256)
seg_norm_kernel(const int* __restrict__ node_idx, int nnz,
                const float* __restrict__ segsum, float* __restrict__ ex)
{
    int t = blockIdx.x * 256 + threadIdx.x;
    if (t >= nnz * 8) return;
    int nz = t >> 3, h = t & 7;
    ex[t] /= segsum[(size_t)node_idx[nz] * 8 + h];
}

__global__ void __launch_bounds__(256)
att1_scatter_kernel(const float* __restrict__ v, const int* __restrict__ node_idx,
                    const int* __restrict__ src_idx, const float* __restrict__ alpha,
                    float* __restrict__ att1)
{
    const int nz = blockIdx.x;
    const int d  = threadIdx.x;
    const int node = node_idx[nz];
    const int src  = src_idx[nz];
    const float w = alpha[(size_t)nz * 8 + (d >> 5)];
    atomicAdd(&att1[(size_t)node * DIMD + d], w * v[(size_t)src * DIMD + d]);
}

// ---------------------------------------------------------------------------
// Host orchestration
// ---------------------------------------------------------------------------
extern "C" void kernel_launch(void* const* d_in, const int* in_sizes, int n_in,
                              void* d_out, int out_size, void* d_ws, size_t ws_size,
                              hipStream_t stream)
{
    const float* x_v    = (const float*)d_in[0];
    const float* x_e    = (const float*)d_in[1];
    const int*   orders = (const int*)d_in[2];
    const int*   nodei  = (const int*)d_in[3];
    const int*   srci   = (const int*)d_in[4];
    const float* pe1    = (const float*)d_in[5];
    const float* pe2    = (const float*)d_in[6];
    const float* peq    = (const float*)d_in[7];
    const float* q_w1   = (const float*)d_in[8];
    const float* q_b1   = (const float*)d_in[9];
    const float* q_w2   = (const float*)d_in[10];
    const float* q_b2   = (const float*)d_in[11];
    const float* k_w    = (const float*)d_in[12];
    const float* k_b    = (const float*)d_in[13];
    const float* v_w    = (const float*)d_in[14];
    const float* v_b    = (const float*)d_in[15];
    const float* m1_w1  = (const float*)d_in[16];
    const float* m1_b1  = (const float*)d_in[17];
    const float* m1_w2  = (const float*)d_in[18];
    const float* m1_b2  = (const float*)d_in[19];
    const float* m2_w1  = (const float*)d_in[20];
    const float* m2_b1  = (const float*)d_in[21];
    const float* m2_w2  = (const float*)d_in[22];
    const float* m2_b2  = (const float*)d_in[23];
    const float* m3_w1  = (const float*)d_in[24];
    const float* m3_b1  = (const float*)d_in[25];
    const float* m3_w2  = (const float*)d_in[26];
    const float* m3_b2  = (const float*)d_in[27];
    const float* ln1_g  = (const float*)d_in[28];
    const float* ln1_b  = (const float*)d_in[29];
    const float* ln2_g  = (const float*)d_in[30];
    const float* ln2_b  = (const float*)d_in[31];
    const float* ln3_g  = (const float*)d_in[32];
    const float* ln3_b  = (const float*)d_in[33];
    const float* bias_b = (const float*)d_in[34];

    const int Nn  = in_sizes[0] / DIMD;   // 50000
    const int E   = in_sizes[1] / DIMD;   // 100000
    const int R   = Nn + E;               // 150000  (multiple of 16)
    const int nnz = in_sizes[3];          // 450000

    // ---- bump allocator over d_ws ----
    char*  ws  = (char*)d_ws;
    size_t off = 0;
    auto alloc = [&](size_t bytes) -> void* {
        void* p = ws + off;
        off += (bytes + 255) & ~(size_t)255;
        return p;
    };

    _Float16* buf1   = (_Float16*)alloc((size_t)R * HIDW * 2);  // cat1 / vbuf alias / cat3+hid3
    _Float16* buf2   = (_Float16*)alloc((size_t)R * HIDW * 2);  // hid1 / cat2+hid2
    _Float16* xs_h   = (_Float16*)alloc((size_t)R * DIMD * 2);
    float*    logits = (float*)alloc((size_t)R * 16 * 4);
    float*    exbuf  = (float*)alloc((size_t)nnz * 8 * 4);
    unsigned* segkey = (unsigned*)alloc((size_t)Nn * 8 * 4);
    float*    segsum = (float*)alloc((size_t)Nn * 8 * 4);
    float*    att1   = (float*)alloc((size_t)Nn * DIMD * 4);
    float*    xbuf   = (float*)alloc((size_t)Nn * DIMD * 4);
    _Float16* m1w1t  = (_Float16*)alloc(HIDW * HIDW * 2);
    _Float16* m1w2t  = (_Float16*)alloc(HIDW * DIMD * 2);
    _Float16* vwt    = (_Float16*)alloc(DIMD * DIMD * 2);
    _Float16* m2w1t  = (_Float16*)alloc(HIDW * HIDW * 2);
    _Float16* m2w2t  = (_Float16*)alloc(HIDW * DIMD * 2);
    _Float16* m3w1t  = (_Float16*)alloc(DIMD * HIDW * 2);
    _Float16* m3w2t  = (_Float16*)alloc(HIDW * DIMD * 2);
    _Float16* wlogt  = (_Float16*)alloc(16 * DIMD * 2);
    float*    qbuf   = (float*)alloc(2 * DIMD * 4);
    float*    blog   = (float*)alloc(16 * 4);
    const int NB = 1024;
    float*    pmax    = (float*)alloc(NB * 8 * 4);
    float*    psum    = (float*)alloc(NB * 8 * 4);
    float*    pacc    = (float*)alloc((size_t)NB * DIMD * 4);
    float*    gmax    = (float*)alloc(8 * 4);
    float*    att0raw = (float*)alloc(DIMD * 4);
    float*    att0fin = (float*)alloc(DIMD * 4);

    // buffer aliases (lifetime-disjoint reuse of buf1/buf2)
    float*    vbuf = (float*)buf1;                     // [R,256] f32
    _Float16* cat2 = buf2;                             // [Nn,512]
    _Float16* hid2 = buf2 + (size_t)Nn * HIDW;         // [Nn,512]
    _Float16* cat3 = buf1;                             // [Nn,256]
    _Float16* hid3 = buf1 + (size_t)Nn * DIMD;         // [Nn,512]

    auto gemm = [&](const _Float16* A, const _Float16* Bt, int M, int K, int Ncol,
                    const float* b1, const float* b2,
                    const float* rA, const float* rB, int split,
                    bool relu, _Float16* oH, float* oF) {
        if (Ncol == 16) {
            dim3 grid((M + 127) / 128, 1);
            if (relu) gemm_wmma_kernel<1, 1, true><<<grid, 256, 0, stream>>>(
                A, Bt, M, K, Ncol, b1, b2, rA, rB, split, oH, oF);
            else      gemm_wmma_kernel<1, 1, false><<<grid, 256, 0, stream>>>(
                A, Bt, M, K, Ncol, b1, b2, rA, rB, split, oH, oF);
        } else {
            dim3 grid((M + 255) / 256, Ncol / 64);
            if (relu) gemm_wmma_kernel<2, 4, true><<<grid, 256, 0, stream>>>(
                A, Bt, M, K, Ncol, b1, b2, rA, rB, split, oH, oF);
            else      gemm_wmma_kernel<2, 4, false><<<grid, 256, 0, stream>>>(
                A, Bt, M, K, Ncol, b1, b2, rA, rB, split, oH, oF);
        }
    };

    // ---- weight prep (transposed f16) ----
    w2h_t_kernel<<<(HIDW * HIDW + 255) / 256, 256, 0, stream>>>(m1_w1, m1w1t, HIDW, HIDW);
    w2h_t_kernel<<<(HIDW * DIMD + 255) / 256, 256, 0, stream>>>(m1_w2, m1w2t, HIDW, DIMD);
    w2h_t_kernel<<<(DIMD * DIMD + 255) / 256, 256, 0, stream>>>(v_w,   vwt,   DIMD, DIMD);
    w2h_t_kernel<<<(HIDW * HIDW + 255) / 256, 256, 0, stream>>>(m2_w1, m2w1t, HIDW, HIDW);
    w2h_t_kernel<<<(HIDW * DIMD + 255) / 256, 256, 0, stream>>>(m2_w2, m2w2t, HIDW, DIMD);
    w2h_t_kernel<<<(DIMD * HIDW + 255) / 256, 256, 0, stream>>>(m3_w1, m3w1t, DIMD, HIDW);
    w2h_t_kernel<<<(HIDW * DIMD + 255) / 256, 256, 0, stream>>>(m3_w2, m3w2t, HIDW, DIMD);

    // ---- q MLP + fold q into k_w -> logit weights ----
    qmlp_kernel<<<2, 256, 0, stream>>>(peq, q_w1, q_b1, q_w2, q_b2, qbuf);
    wlogit_kernel<<<16, 256, 0, stream>>>(k_w, k_b, qbuf, wlogt, blog);

    // ---- stage 1: per-row m1 MLP on [x_e ; x_v] (edges first, matching xs) ----
    build_cat_kernel<<<R, 256, 0, stream>>>(x_e, x_v, E, orders, 1, pe1, ln1_g, ln1_b, buf1);
    gemm(buf1, m1w1t, R, 2 * DIMD, HIDW, m1_b1, nullptr, nullptr, nullptr, 0, true, buf2, nullptr);
    gemm(buf2, m1w2t, R, HIDW, DIMD, m1_b2, nullptr, x_e, x_v, E, false, xs_h, nullptr);

    // ---- stage 2: V projection + fused logit GEMM (k_all never materialized) ----
    gemm(xs_h, vwt,   R, DIMD, DIMD, v_b,  nullptr, nullptr, nullptr, 0, false, nullptr, vbuf);
    gemm(xs_h, wlogt, R, DIMD, 16,   blog, nullptr, nullptr, nullptr, 0, false, nullptr, logits);

    // ---- stage 3: global (order-0) attention -> att0fin [256] ----
    att0_max_kernel<<<NB, 256, 0, stream>>>(logits, R, pmax);
    att0_maxfin_kernel<<<1, 256, 0, stream>>>(pmax, NB, gmax);
    att0_acc_kernel<<<NB, 256, 0, stream>>>(logits, vbuf, gmax, R, pacc, psum);
    att0_fin_kernel<<<1, 256, 0, stream>>>(pacc, psum, NB, att0raw);
    att0_mlp_kernel<<<1, 256, 0, stream>>>(att0raw, pe2, ln2_g, ln2_b,
                                           m2_w1, m2_b1, m2_w2, m2_b2, att0fin);

    // ---- stage 4: sparse (order-1) segment softmax -> att1 [Nn,256] ----
    hipMemsetAsync(segkey, 0, (size_t)Nn * 8 * 4, stream);   // key 0 < key(-inf)
    hipMemsetAsync(segsum, 0, (size_t)Nn * 8 * 4, stream);
    hipMemsetAsync(att1,   0, (size_t)Nn * DIMD * 4, stream);
    const int g8 = (nnz * 8 + 255) / 256;
    seg_max_kernel<<<g8, 256, 0, stream>>>(logits, nodei, srci, nnz, segkey);
    seg_exp_kernel<<<g8, 256, 0, stream>>>(logits, nodei, srci, nnz, segkey, exbuf, segsum);
    seg_norm_kernel<<<g8, 256, 0, stream>>>(nodei, nnz, segsum, exbuf);
    att1_scatter_kernel<<<nnz, 256, 0, stream>>>(vbuf, nodei, srci, exbuf, att1);

    // ---- stage 5: att1 m2 MLP; x = att0fin + att1_upd (att0fin as 2nd bias) ----
    build_cat_kernel<<<Nn, 256, 0, stream>>>(att1, att1, Nn, nullptr, 1, pe2, ln2_g, ln2_b, cat2);
    gemm(cat2, m2w1t, Nn, 2 * DIMD, HIDW, m2_b1, nullptr, nullptr, nullptr, 0, true, hid2, nullptr);
    gemm(hid2, m2w2t, Nn, HIDW, DIMD, m2_b2, att0fin, att1, att1, Nn, false, nullptr, xbuf);

    // ---- stage 6: final m3 MLP; out = x + mlp(ln3(x)) + bias_b ----
    build_ln_kernel<<<Nn, 256, 0, stream>>>(xbuf, ln3_g, ln3_b, cat3);
    gemm(cat3, m3w1t, Nn, DIMD, HIDW, m3_b1, nullptr, nullptr, nullptr, 0, true, hid3, nullptr);
    gemm(hid3, m3w2t, Nn, HIDW, DIMD, m3_b2, bias_b, xbuf, xbuf, Nn, false, nullptr, (float*)d_out);

    (void)n_in; (void)out_size; (void)ws_size;
}
